// DAWNLayer_10419590660471
// MI455X (gfx1250) — matrix operand order repaired and checked by reading, not compiled, not run on hardware
//
#include <hip/hip_runtime.h>
#include <hip/hip_bf16.h>

// ---------------------------------------------------------------------------
// DAWN layer for gfx1250 (MI455X): WMMA f16 pipeline with async global->LDS
// double buffering everywhere. softmax.mean(-1) == 1/S simplification removes
// the attention dependency from the top-k path.
// ---------------------------------------------------------------------------

typedef __attribute__((ext_vector_type(16))) _Float16 v16h;
typedef __attribute__((ext_vector_type(8)))  _Float16 v8h;
typedef __attribute__((ext_vector_type(8)))  float    v8f;

#define TOKENS 8192   // B*S
#define DMODEL 768
#define SEQ    1024
#define NHEAD  12
#define DHEAD  64
#define DFF    3072
#define NNEUR  96
#define SEMLD  128    // semantic projection padded to 128 cols
#define NBASIS 32
#define RANK   64
#define TOPK   8

__device__ __forceinline__ v8f wmma16(v16h a, v16h b, v8f c) {
  return __builtin_amdgcn_wmma_f32_16x16x32_f16(false, a, false, b, (short)0, c,
                                                false, false);
}

// Async 16-byte global->LDS copy (CDNA5, ASYNCcnt-tracked).
__device__ __forceinline__ void async_b128(unsigned int lds_off, const void* g) {
  asm volatile("global_load_async_to_lds_b128 %0, %1, off"
               :: "v"(lds_off), "v"((unsigned long long)(size_t)g)
               : "memory");
}
__device__ __forceinline__ void wait_async0() {
  asm volatile("s_wait_asynccnt 0x0" ::: "memory");
}
__device__ __forceinline__ unsigned int lds_addr(const void* p) {
  return (unsigned int)(size_t)p;
}

// A-matrix fragment (16x32 f16), ISA 7.12.2 layout.
__device__ __forceinline__ v16h load_a_frag(const _Float16* base, int stride,
                                            int lane) {
  int r  = lane & 15;
  int kb = (lane >> 4) << 3;
  v8h lo = *(const v8h*)(base + r * stride + kb);
  v8h hi = *(const v8h*)(base + r * stride + 16 + kb);
  v16h a;
#pragma unroll
  for (int i = 0; i < 8; ++i) { a[i] = lo[i]; a[i + 8] = hi[i]; }
  return a;
}

// B-matrix fragment (32x16 f16): col n = cb + lane%16, halfs = K rows in table order.
__device__ __forceinline__ v16h load_b_frag(const _Float16* base, int stride,
                                            int lane, int cb) {
  int n  = lane & 15;
  int kb = (lane >> 4) << 3;
  v16h b;
#pragma unroll
  for (int i = 0; i < 8; ++i) {
    b[i] = base[(kb + i) * stride + cb + n];
    b[i + 8] = base[(16 + kb + i) * stride + cb + n];
  }
  return b;
}

__device__ __forceinline__ float red16_max(float v) {
#pragma unroll
  for (int m = 1; m <= 8; m <<= 1) v = fmaxf(v, __shfl_xor(v, m, 32));
  return v;
}
__device__ __forceinline__ float red16_sum(float v) {
#pragma unroll
  for (int m = 1; m <= 8; m <<= 1) v += __shfl_xor(v, m, 32);
  return v;
}

// ---------------------------------------------------------------------------
// f32 -> f16 conversion
// ---------------------------------------------------------------------------
__global__ void k_cvt_f16(const float* __restrict__ in, _Float16* __restrict__ out,
                          int n) {
  int i = blockIdx.x * blockDim.x + threadIdx.x;
  if (i < n) out[i] = (_Float16)in[i];
}

// ---------------------------------------------------------------------------
// LayerNorm -> f16
// ---------------------------------------------------------------------------
__global__ void k_layernorm(const float* __restrict__ x, const float* __restrict__ g,
                            const float* __restrict__ b, _Float16* __restrict__ out) {
  int row = blockIdx.x;
  const float* xr = x + (size_t)row * DMODEL;
  __shared__ float r1[256], r2[256];
  int t = threadIdx.x;
  float s = 0.f, s2 = 0.f;
  for (int i = t; i < DMODEL; i += 256) { float v = xr[i]; s += v; s2 += v * v; }
  r1[t] = s; r2[t] = s2;
  __syncthreads();
  for (int o = 128; o > 0; o >>= 1) {
    if (t < o) { r1[t] += r1[t + o]; r2[t] += r2[t + o]; }
    __syncthreads();
  }
  float mean = r1[0] * (1.f / DMODEL);
  float var  = r2[0] * (1.f / DMODEL) - mean * mean;
  float rstd = rsqrtf(var + 1e-5f);
  for (int i = t; i < DMODEL; i += 256) {
    float v = (xr[i] - mean) * rstd * g[i] + b[i];
    out[(size_t)row * DMODEL + i] = (_Float16)v;
  }
}

// ---------------------------------------------------------------------------
// Prep: recipe softmax, neuron_emb^T padded to [768, SEMLD] f16, sigmoid
// context consts.  context[n] = sigmoid( sum_h ctx_pat[n,h] / S )
// ---------------------------------------------------------------------------
__global__ void k_prep(const float* __restrict__ recipe,
                       const float* __restrict__ ctx_pat,
                       const float* __restrict__ basis_emb,
                       float* __restrict__ rnorm_g,
                       float* __restrict__ ctx_sig,
                       _Float16* __restrict__ bsemT) {
  __shared__ float rn[NNEUR * NBASIS];
  int t = threadIdx.x;  // 128
  if (t < NNEUR) {
    const float* r = recipe + t * NBASIS;
    float mx = -1e30f;
#pragma unroll
    for (int j = 0; j < NBASIS; ++j) mx = fmaxf(mx, r[j]);
    float e[NBASIS], den = 0.f;
#pragma unroll
    for (int j = 0; j < NBASIS; ++j) { e[j] = __expf(r[j] - mx); den += e[j]; }
    float inv = 1.f / den;
#pragma unroll
    for (int j = 0; j < NBASIS; ++j) {
      float v = e[j] * inv;
      rn[t * NBASIS + j] = v;
      rnorm_g[t * NBASIS + j] = v;
    }
    float c = 0.f;
#pragma unroll
    for (int h = 0; h < NHEAD; ++h) c += ctx_pat[t * NHEAD + h];
    c *= (1.f / (float)SEQ);
    ctx_sig[t] = 1.f / (1.f + __expf(-c));
  }
  __syncthreads();
  for (int i = t; i < DMODEL * SEMLD; i += 128) {
    int k = i >> 7, n = i & (SEMLD - 1);
    float acc = 0.f;
    if (n < NNEUR) {
#pragma unroll
      for (int j = 0; j < NBASIS; ++j)
        acc += rn[n * NBASIS + j] * basis_emb[j * DMODEL + k];
    }
    bsemT[i] = (_Float16)acc;
  }
}

// ---------------------------------------------------------------------------
// WMMA GEMM: C[M,N] = act(A[M,K]@B[K,N] + bias) + resid
// BM=128 BN=128 BK=32, 256 threads = 8 waves; async double-buffered LDS;
// last k-step peeled so the steady-state loop is branch-free; epilogue fully
// specialized by template parameters.  M%128==0, N%128==0, K%32==0.
// ---------------------------------------------------------------------------
#define AS_STR 40
#define BS_STR 136
template <bool HAS_BIAS, int ACT, bool HAS_RES, bool OUT_F, bool OUT_H>
__global__ void k_gemm(const _Float16* __restrict__ A, const _Float16* __restrict__ B,
                       const float* __restrict__ bias, const float* __restrict__ resid,
                       float* __restrict__ outF, _Float16* __restrict__ outH,
                       int M, int N, int Kd) {
  __shared__ __align__(16) _Float16 As[2][128 * AS_STR];
  __shared__ __align__(16) _Float16 Bs[2][32 * BS_STR];
  const int tn = blockIdx.x * 128, tm = blockIdx.y * 128;
  const int wave = threadIdx.x >> 5, lane = threadIdx.x & 31;
  const int ar = threadIdx.x >> 1, ac = (threadIdx.x & 1) * 16;   // A: 128x32
  const int br = threadIdx.x >> 3, bc = (threadIdx.x & 7) * 16;   // B: 32x128

  v8f acc[8];
#pragma unroll
  for (int i = 0; i < 8; ++i) acc[i] = (v8f)0.0f;

  auto issue = [&](int buf, int k0) {
    const _Float16* ap = A + (size_t)(tm + ar) * Kd + k0 + ac;
    unsigned int la = lds_addr(&As[buf][ar * AS_STR + ac]);
    async_b128(la,      ap);
    async_b128(la + 16, ap + 8);
    const _Float16* bp = B + (size_t)(k0 + br) * N + tn + bc;
    unsigned int lb = lds_addr(&Bs[buf][br * BS_STR + bc]);
    async_b128(lb,      bp);
    async_b128(lb + 16, bp + 8);
  };

  // compute one k-step: preload all 8 B-frags so the 8 independent WMMAs
  // issue back-to-back after a single DS wait (XDL pipelining).
  auto compute = [&](int cur) {
    v16h a = load_a_frag(&As[cur][wave * 16 * AS_STR], AS_STR, lane);
    v16h bfr[8];
#pragma unroll
    for (int nt = 0; nt < 8; ++nt)
      bfr[nt] = load_b_frag(&Bs[cur][0], BS_STR, lane, nt * 16);
#pragma unroll
    for (int nt = 0; nt < 8; ++nt) acc[nt] = wmma16(a, bfr[nt], acc[nt]);
  };

  issue(0, 0);
  wait_async0();
  __syncthreads();

  int k0 = 0;
  for (; k0 < Kd - 32; k0 += 32) {
    const int cur = (k0 >> 5) & 1;
    issue(cur ^ 1, k0 + 32);
    if (k0 + 64 < Kd) {  // warm GL2 ahead of the async engine
      __builtin_prefetch(A + (size_t)(tm + ar) * Kd + k0 + 64 + ac, 0, 3);
      __builtin_prefetch(B + (size_t)(k0 + 64 + br) * N + tn + bc, 0, 3);
    }
    compute(cur);
    wait_async0();
    __syncthreads();
  }
  compute((k0 >> 5) & 1);  // peeled last k-step

  const int rbase = tm + wave * 16 + ((lane >> 4) << 3);
  const int cl = lane & 15;
#pragma unroll
  for (int nt = 0; nt < 8; ++nt) {
    int gc = tn + nt * 16 + cl;
    float bv = HAS_BIAS ? bias[gc] : 0.f;
#pragma unroll
    for (int e = 0; e < 8; ++e) {
      int gr = rbase + e;
      float v = acc[nt][e] + bv;
      if (ACT == 1) v = 0.5f * v * (1.f + erff(v * 0.70710678f));
      if (HAS_RES) v += resid[(size_t)gr * N + gc];
      if (OUT_F) outF[(size_t)gr * N + gc] = v;
      if (OUT_H) outH[(size_t)gr * N + gc] = (_Float16)v;
    }
  }
}

// ---------------------------------------------------------------------------
// top-8 neurons + recipe mixture: one thread per token.
// ---------------------------------------------------------------------------
__global__ void k_topk(const float* __restrict__ sem, const float* __restrict__ ctx_sig,
                       const float* __restrict__ rnorm, float* __restrict__ tr) {
  int tok = blockIdx.x * blockDim.x + threadIdx.x;
  if (tok >= TOKENS) return;
  const float* s = sem + (size_t)tok * SEMLD;
  int   idx[TOPK];
  float sc[TOPK];
#pragma unroll
  for (int k = 0; k < TOPK; ++k) {
    float best = -1e30f; int bi = 0;
    for (int n = 0; n < NNEUR; ++n) {
      float v = s[n] * ctx_sig[n];
      bool used = false;
#pragma unroll
      for (int j = 0; j < TOPK; ++j) used |= (j < k) && (idx[j] == n);
      if (!used && v > best) { best = v; bi = n; }
    }
    idx[k] = bi; sc[k] = best;
  }
  float mx = sc[0];
#pragma unroll
  for (int k = 1; k < TOPK; ++k) mx = fmaxf(mx, sc[k]);
  float w[TOPK], den = 0.f;
#pragma unroll
  for (int k = 0; k < TOPK; ++k) { w[k] = __expf(sc[k] - mx); den += w[k]; }
  float inv = 1.f / den;
  float accv[NBASIS];
#pragma unroll
  for (int j = 0; j < NBASIS; ++j) accv[j] = 0.f;
#pragma unroll
  for (int k = 0; k < TOPK; ++k) {
    const float* r = rnorm + idx[k] * NBASIS;
    float wk = w[k] * inv;
#pragma unroll
    for (int j = 0; j < NBASIS; ++j) accv[j] += wk * r[j];
  }
#pragma unroll
  for (int j = 0; j < NBASIS; ++j) tr[(size_t)tok * NBASIS + j] = accv[j];
}

// ---------------------------------------------------------------------------
// v_sem[t,r] = sum_n tr[t,n] * (normed[t,:] @ basis_A[n,:,r])
// 4 waves, 64-token tile. The whole 64x768 activation strip is staged in LDS
// once via async copies (96 KB of the 320 KB WGP LDS); basis tiles stream
// through a double-buffered async pipeline over the flattened (n, k) loop.
// ---------------------------------------------------------------------------
#define VA_STR 776   // 768 + 8 halfs padding
__global__ void k_vsem(const _Float16* __restrict__ Ah, const _Float16* __restrict__ basisA,
                       const float* __restrict__ tr, _Float16* __restrict__ vout) {
  __shared__ __align__(16) _Float16 As[64 * VA_STR];
  __shared__ __align__(16) _Float16 Bs[2][32 * 72];
  const int tm = blockIdx.x * 64;
  const int wave = threadIdx.x >> 5, lane = threadIdx.x & 31;
  const int rloc = wave * 16 + ((lane >> 4) << 3);
  const int KT = DMODEL / 32, TOT = NBASIS * KT;

  auto issueB = [&](int buf, int idx) {
    int n = idx / KT, kt = idx % KT;
    int row = threadIdx.x >> 2, off = (threadIdx.x & 3) * 16;
    const _Float16* bp = basisA + (size_t)n * DMODEL * RANK +
                         (size_t)(kt * 32 + row) * RANK + off;
    unsigned int lb = lds_addr(&Bs[buf][row * 72 + off]);
    async_b128(lb,      bp);
    async_b128(lb + 16, bp + 8);
  };

  // stage the 64x768 activation strip (async) + first basis tile
  for (int i = threadIdx.x; i < 64 * 48; i += 128) {
    int row = i / 48, ch = (i % 48) * 16;
    const _Float16* ap = Ah + (size_t)(tm + row) * DMODEL + ch;
    unsigned int la = lds_addr(&As[row * VA_STR + ch]);
    async_b128(la,      ap);
    async_b128(la + 16, ap + 8);
  }
  issueB(0, 0);
  wait_async0();
  __syncthreads();

  v8f tot[4], acc[4];
#pragma unroll
  for (int i = 0; i < 4; ++i) { tot[i] = (v8f)0.0f; acc[i] = (v8f)0.0f; }

  for (int idx = 0; idx < TOT; ++idx) {
    const int cur = idx & 1;
    const bool has_next = (idx + 1) < TOT;
    if (has_next) issueB(cur ^ 1, idx + 1);

    const int n = idx / KT, kt = idx % KT;
    v16h a = load_a_frag(&As[wave * 16 * VA_STR + kt * 32], VA_STR, lane);
    v16h bfr[4];
#pragma unroll
    for (int nt = 0; nt < 4; ++nt)
      bfr[nt] = load_b_frag(&Bs[cur][0], 72, lane, nt * 16);
#pragma unroll
    for (int nt = 0; nt < 4; ++nt) acc[nt] = wmma16(a, bfr[nt], acc[nt]);

    if (kt == KT - 1) {  // end of basis n: fold weighted partial into total
      float wr[8];
#pragma unroll
      for (int e = 0; e < 8; ++e)
        wr[e] = tr[(size_t)(tm + rloc + e) * NBASIS + n];
#pragma unroll
      for (int nt = 0; nt < 4; ++nt)
#pragma unroll
        for (int e = 0; e < 8; ++e) {
          tot[nt][e] += wr[e] * acc[nt][e];
          acc[nt][e] = 0.f;
        }
    }
    if (has_next) wait_async0();
    __syncthreads();
  }

  int cl = lane & 15;
#pragma unroll
  for (int nt = 0; nt < 4; ++nt)
#pragma unroll
    for (int e = 0; e < 8; ++e)
      vout[(size_t)(tm + rloc + e) * RANK + nt * 16 + cl] = (_Float16)tot[nt][e];
}

// ---------------------------------------------------------------------------
// Flash attention per (b, h, 64-query block). 128 threads = 4 waves; each wave
// owns 16 query rows. 32-key blocks stream through async double-buffered LDS;
// QK^T and P@V via WMMA; P round-trips through LDS for layout conversion.
// ---------------------------------------------------------------------------
__global__ void k_flash(const _Float16* __restrict__ Qh, const _Float16* __restrict__ Kh,
                        const _Float16* __restrict__ Vh, _Float16* __restrict__ Oh) {
  const int h = blockIdx.y, b = blockIdx.z;
  const int qbase = b * SEQ + blockIdx.x * 64;
  const int hc = h * DHEAD;
  const int wave = threadIdx.x >> 5, lane = threadIdx.x & 31;

  __shared__ __align__(16) _Float16 Ks[2][32 * 72];
  __shared__ __align__(16) _Float16 Vs[2][32 * 72];
  __shared__ __align__(16) _Float16 Ps[4][16 * 40];

  auto issueKV = [&](int buf, int t0) {
    int row = threadIdx.x >> 2, ch = (threadIdx.x & 3) * 16;
    const _Float16* kp = Kh + (size_t)(b * SEQ + t0 + row) * DMODEL + hc + ch;
    const _Float16* vp = Vh + (size_t)(b * SEQ + t0 + row) * DMODEL + hc + ch;
    unsigned int lk = lds_addr(&Ks[buf][row * 72 + ch]);
    unsigned int lv = lds_addr(&Vs[buf][row * 72 + ch]);
    async_b128(lk,      kp);
    async_b128(lk + 16, kp + 8);
    async_b128(lv,      vp);
    async_b128(lv + 16, vp + 8);
  };

  const int qrow = qbase + wave * 16;
  v16h qa[2];
  {
    int r = lane & 15, kb = (lane >> 4) << 3;
    const _Float16* qp = Qh + (size_t)(qrow + r) * DMODEL + hc;
#pragma unroll
    for (int c = 0; c < 2; ++c) {
      v8h lo = *(const v8h*)(qp + c * 32 + kb);
      v8h hi = *(const v8h*)(qp + c * 32 + 16 + kb);
#pragma unroll
      for (int i = 0; i < 8; ++i) { qa[c][i] = lo[i]; qa[c][i + 8] = hi[i]; }
    }
  }

  float m[8], l[8];
  v8f o[4];
#pragma unroll
  for (int e = 0; e < 8; ++e) { m[e] = -1e30f; l[e] = 0.f; }
#pragma unroll
  for (int i = 0; i < 4; ++i) o[i] = (v8f)0.0f;

  issueKV(0, 0);
  wait_async0();
  __syncthreads();

  for (int t0 = 0; t0 < SEQ; t0 += 32) {
    const int cur = (t0 >> 5) & 1;
    const bool has_next = (t0 + 32) < SEQ;
    if (has_next) issueKV(cur ^ 1, t0 + 32);

    v8f sf[2];
#pragma unroll
    for (int tt = 0; tt < 2; ++tt) {
      v8f s = (v8f)0.0f;
#pragma unroll
      for (int c = 0; c < 2; ++c) {
        v16h bb;  // B[k=d, n=key]: lane%16 = key col, halfs = d in table order
#pragma unroll
        for (int i = 0; i < 16; ++i)
          bb[i] = Ks[cur][(tt * 16 + (lane & 15)) * 72 + c * 32 +
                          ((lane >> 4) << 3) + (i & 7) + ((i >> 3) << 4)];
        s = wmma16(qa[c], bb, s);
      }
#pragma unroll
      for (int e = 0; e < 8; ++e) s[e] *= 0.125f;  // 1/sqrt(64)
      sf[tt] = s;
    }

    float mnew[8], alpha[8], lsum[8];
#pragma unroll
    for (int e = 0; e < 8; ++e) {
      float mx = red16_max(fmaxf(sf[0][e], sf[1][e]));
      mnew[e]  = fmaxf(m[e], mx);
      alpha[e] = __expf(m[e] - mnew[e]);
      lsum[e]  = 0.f;
    }
    int rr = (lane >> 4) << 3, cc = lane & 15;
#pragma unroll
    for (int tt = 0; tt < 2; ++tt)
#pragma unroll
      for (int e = 0; e < 8; ++e) {
        float p = __expf(sf[tt][e] - mnew[e]);
        lsum[e] += p;
        Ps[wave][(rr + e) * 40 + tt * 16 + cc] = (_Float16)p;
      }
#pragma unroll
    for (int e = 0; e < 8; ++e) {
      l[e] = l[e] * alpha[e] + red16_sum(lsum[e]);
      m[e] = mnew[e];
    }
#pragma unroll
    for (int nt = 0; nt < 4; ++nt)
#pragma unroll
      for (int e = 0; e < 8; ++e) o[nt][e] *= alpha[e];

    v16h pa = load_a_frag(&Ps[wave][0], 40, lane);
    v16h vb[4];
#pragma unroll
    for (int nt = 0; nt < 4; ++nt)
      vb[nt] = load_b_frag(&Vs[cur][0], 72, lane, nt * 16);
#pragma unroll
    for (int nt = 0; nt < 4; ++nt) o[nt] = wmma16(pa, vb[nt], o[nt]);

    if (has_next) wait_async0();
    __syncthreads();
  }

  int rr = (lane >> 4) << 3, cc = lane & 15;
#pragma unroll
  for (int nt = 0; nt < 4; ++nt)
#pragma unroll
    for (int e = 0; e < 8; ++e) {
      float v = o[nt][e] / l[e];
      Oh[(size_t)(qrow + rr + e) * DMODEL + hc + nt * 16 + cc] = (_Float16)v;
    }
}

// ---------------------------------------------------------------------------
// Host-side orchestration
// ---------------------------------------------------------------------------
extern "C" void kernel_launch(void* const* d_in, const int* in_sizes, int n_in,
                              void* d_out, int out_size, void* d_ws, size_t ws_size,
                              hipStream_t stream) {
  const float* x        = (const float*)d_in[0];
  const float* q_w      = (const float*)d_in[1];
  const float* q_b      = (const float*)d_in[2];
  const float* k_w      = (const float*)d_in[3];
  const float* k_b      = (const float*)d_in[4];
  const float* ao_w     = (const float*)d_in[5];
  const float* ao_b     = (const float*)d_in[6];
  const float* recipe   = (const float*)d_in[7];
  const float* ctx_pat  = (const float*)d_in[8];
  const float* basis_e  = (const float*)d_in[9];
  const float* basis_A  = (const float*)d_in[10];
  const float* vout_w   = (const float*)d_in[11];
  const float* vout_b   = (const float*)d_in[12];
  const float* up_w     = (const float*)d_in[13];
  const float* up_b     = (const float*)d_in[14];
  const float* down_w   = (const float*)d_in[15];
  const float* down_b   = (const float*)d_in[16];
  const float* n1_g     = (const float*)d_in[17];
  const float* n1_b     = (const float*)d_in[18];
  const float* n2_g     = (const float*)d_in[19];
  const float* n2_b     = (const float*)d_in[20];
  float* out = (float*)d_out;

  char* W = (char*)d_ws;
  size_t off = 0;
  auto take = [&](size_t bytes) { size_t o = off; off += (bytes + 255) & ~(size_t)255; return o; };
  const size_t T = TOKENS;
  size_t o_normed = take(T * DMODEL * 2);
  size_t o_qw     = take((size_t)DMODEL * DMODEL * 2);
  size_t o_kw     = take((size_t)DMODEL * DMODEL * 2);
  size_t o_aow    = take((size_t)DMODEL * DMODEL * 2);
  size_t o_vow    = take((size_t)RANK * DMODEL * 2);
  size_t o_upw    = take((size_t)DMODEL * DFF * 2);
  size_t o_dnw    = take((size_t)DFF * DMODEL * 2);
  size_t o_bA     = take((size_t)NBASIS * DMODEL * RANK * 2);
  size_t o_bsem   = take((size_t)DMODEL * SEMLD * 2);
  size_t o_rn     = take((size_t)NNEUR * NBASIS * 4);
  size_t o_csig   = take((size_t)NNEUR * 4);
  size_t o_Q      = take(T * DMODEL * 2);
  size_t o_K      = take(T * DMODEL * 2);
  size_t o_V      = take(T * DMODEL * 2);
  size_t o_sem    = take(T * SEMLD * 4);
  size_t o_tr     = take(T * NBASIS * 4);
  size_t o_vsem   = take(T * RANK * 2);
  size_t o_attn   = take(T * DMODEL * 2);
  size_t o_x2     = take(T * DMODEL * 4);
  size_t o_n2     = take(T * DMODEL * 2);
  size_t o_ff     = take(T * DFF * 2);

  auto H = [&](size_t o) { return (_Float16*)(W + o); };
  auto F = [&](size_t o) { return (float*)(W + o); };

  auto cvt = [&](const float* src, _Float16* dst, int n) {
    k_cvt_f16<<<(n + 255) / 256, 256, 0, stream>>>(src, dst, n);
  };
  cvt(q_w,    H(o_qw),  DMODEL * DMODEL);
  cvt(k_w,    H(o_kw),  DMODEL * DMODEL);
  cvt(ao_w,   H(o_aow), DMODEL * DMODEL);
  cvt(vout_w, H(o_vow), RANK * DMODEL);
  cvt(up_w,   H(o_upw), DMODEL * DFF);
  cvt(down_w, H(o_dnw), DFF * DMODEL);
  cvt(basis_A, H(o_bA), NBASIS * DMODEL * RANK);

  k_prep<<<1, 128, 0, stream>>>(recipe, ctx_pat, basis_e, F(o_rn), F(o_csig), H(o_bsem));
  k_layernorm<<<TOKENS, 256, 0, stream>>>(x, n1_g, n1_b, H(o_normed));

  // GEMM config aliases:
  //   bias->f16 out            (Q, K, V-out)
  //   no-bias->f32 out         (semantic)
  //   bias+resid->f32 out      (AO, FFN-down)
  //   bias+GELU->f16 out       (FFN-up)
  dim3 blk(256);
  {
    dim3 g(DMODEL / 128, TOKENS / 128);
    k_gemm<true, 0, false, false, true><<<g, blk, 0, stream>>>(
        H(o_normed), H(o_qw), q_b, nullptr, nullptr, H(o_Q), TOKENS, DMODEL, DMODEL);
    k_gemm<true, 0, false, false, true><<<g, blk, 0, stream>>>(
        H(o_normed), H(o_kw), k_b, nullptr, nullptr, H(o_K), TOKENS, DMODEL, DMODEL);
  }
  {
    dim3 g(SEMLD / 128, TOKENS / 128);
    k_gemm<false, 0, false, true, false><<<g, blk, 0, stream>>>(
        H(o_normed), H(o_bsem), nullptr, nullptr, F(o_sem), nullptr, TOKENS, SEMLD, DMODEL);
  }

  k_topk<<<TOKENS / 256, 256, 0, stream>>>(F(o_sem), F(o_csig), F(o_rn), F(o_tr));
  k_vsem<<<TOKENS / 64, 128, 0, stream>>>(H(o_normed), H(o_bA), F(o_tr), H(o_vsem));

  {
    dim3 g(DMODEL / 128, TOKENS / 128);
    k_gemm<true, 0, false, false, true><<<g, blk, 0, stream>>>(
        H(o_vsem), H(o_vow), vout_b, nullptr, nullptr, H(o_V), TOKENS, DMODEL, RANK);
  }
  {
    dim3 g(SEQ / 64, NHEAD, 8);
    k_flash<<<g, 128, 0, stream>>>(H(o_Q), H(o_K), H(o_V), H(o_attn));
  }
  {
    dim3 g(DMODEL / 128, TOKENS / 128);
    k_gemm<true, 0, true, true, false><<<g, blk, 0, stream>>>(
        H(o_attn), H(o_aow), ao_b, x, F(o_x2), nullptr, TOKENS, DMODEL, DMODEL);
  }
  k_layernorm<<<TOKENS, 256, 0, stream>>>(F(o_x2), n2_g, n2_b, H(o_n2));
  {
    dim3 g(DFF / 128, TOKENS / 128);
    k_gemm<true, 1, false, false, true><<<g, blk, 0, stream>>>(
        H(o_n2), H(o_upw), up_b, nullptr, nullptr, H(o_ff), TOKENS, DFF, DMODEL);
  }
  {
    dim3 g(DMODEL / 128, TOKENS / 128);
    k_gemm<true, 0, true, true, false><<<g, blk, 0, stream>>>(
        H(o_ff), H(o_dnw), down_b, F(o_x2), out, nullptr, TOKENS, DMODEL, DFF);
  }
}
